// MHNA_5660766896672
// MI455X (gfx1250) — compile-verified, matrix operand order checked by reading, not run on hardware
//
#include <hip/hip_runtime.h>

#define D_MODEL   1024
#define NUM_HEADS 16
#define HEAD_DIM  64
#define BATCH     2
#define SEQ       2048
#define ROWS      (BATCH * SEQ)   // 4096

typedef __bf16 bf16_t;
typedef __attribute__((ext_vector_type(16))) __bf16        v16bf;
typedef __attribute__((ext_vector_type(8)))  float         v8f;
typedef __attribute__((ext_vector_type(4)))  unsigned int  uint4v;

// 16-bit operand K-offset mapping for V_WMMA_F32_16X16X32_BF16 (ISA 7.12.2):
// lanes 0-15: VGPR0-3 -> K 0..7, VGPR4-7 -> K 16..23 ; lanes 16-31: +8
__device__ __forceinline__ int koff16(int v, int half, int slot) {
  return ((v & 3) << 1) + ((v >> 2) << 4) + (half << 3) + slot;
}

// ---------------------------------------------------------------------------
// f32 -> bf16 conversion
// ---------------------------------------------------------------------------
__global__ void f32_to_bf16(const float* __restrict__ in,
                            bf16_t* __restrict__ out, int n) {
  int i = blockIdx.x * blockDim.x + threadIdx.x;
  if (i < n) out[i] = (bf16_t)in[i];
}

// ---------------------------------------------------------------------------
// n = x @ Wn + bn   (M=4096, N=16, K=1024) — tiny, f32 scalar
// ---------------------------------------------------------------------------
__global__ void proj_n(const float* __restrict__ x, const float* __restrict__ Wn,
                       const float* __restrict__ bn, float* __restrict__ nbuf) {
  int idx = blockIdx.x * blockDim.x + threadIdx.x;  // ROWS*NUM_HEADS
  if (idx >= ROWS * NUM_HEADS) return;
  int row = idx >> 4, h = idx & 15;
  const float* xr = x + (long)row * D_MODEL;
  float acc = bn[h];
  for (int k = 0; k < D_MODEL; ++k) acc += xr[k] * Wn[k * NUM_HEADS + h];
  nbuf[idx] = acc;
}

// ---------------------------------------------------------------------------
// C[M,N] = A[M,K](bf16) @ B[K,N](bf16) + bias[N]   via WMMA bf16->f32
// block tile 128x128, 8 waves (4 along M x 2 along N), wave tile 32x64
// Double-buffered LDS: next k-tile's global_load_b128 issued before compute.
// ---------------------------------------------------------------------------
template <bool OUT_BF16>
__global__ __launch_bounds__(256) void gemm_bias_wmma(
    const bf16_t* __restrict__ A, const bf16_t* __restrict__ B,
    const float* __restrict__ bias, void* __restrict__ C,
    int M, int N, int K) {
  __shared__ bf16_t sA[2][128 * 40];   // 128 rows x 32 k (pad to 40)
  __shared__ bf16_t sB[2][32 * 136];   // 32 k rows x 128 n (pad to 136)

  const int tid = threadIdx.x;
  const int lane = tid & 31;
  // scalarized wave id: keeps wave-level control flow in SALU (WMMA needs
  // full EXEC; no divergent branches may surround matrix ops)
  const int wave = __builtin_amdgcn_readfirstlane(tid >> 5);
  const int wm = wave & 3, wn = wave >> 2;
  const int half = lane >> 4, l16 = lane & 15;
  const long bm = (long)blockIdx.x * 128;
  const long bn = (long)blockIdx.y * 128;

  // two b128 chunks per tile per thread (A: 128x32, B: 32x128)
  const int ra0 = tid >> 2,         ca = (tid & 3) << 3;
  const int ra1 = (tid + 256) >> 2;
  const int rb0 = tid >> 4,         cb = (tid & 15) << 3;
  const int rb1 = (tid + 256) >> 4;
  const long arow0 = (bm + ra0) * (long)K;
  const long arow1 = (bm + ra1) * (long)K;

  const v8f vzero = {0.f, 0.f, 0.f, 0.f, 0.f, 0.f, 0.f, 0.f};
  v8f acc[2][4];
#pragma unroll
  for (int mt = 0; mt < 2; ++mt)
#pragma unroll
    for (int nt = 0; nt < 4; ++nt) acc[mt][nt] = vzero;

  // prologue: stage k-tile 0 into buffer 0
  uint4v a0 = *(const uint4v*)&A[arow0 + ca];
  uint4v a1 = *(const uint4v*)&A[arow1 + ca];
  uint4v b0 = *(const uint4v*)&B[(long)rb0 * N + bn + cb];
  uint4v b1 = *(const uint4v*)&B[(long)rb1 * N + bn + cb];
  *(uint4v*)&sA[0][ra0 * 40 + ca] = a0;
  *(uint4v*)&sA[0][ra1 * 40 + ca] = a1;
  *(uint4v*)&sB[0][rb0 * 136 + cb] = b0;
  *(uint4v*)&sB[0][rb1 * 136 + cb] = b1;
  __syncthreads();

  const int nk = K >> 5;
  for (int kt = 0; kt < nk; ++kt) {
    const int cur = kt & 1;
    const bool more = (kt + 1) < nk;
    if (more) {  // issue next tile's loads early; latency hidden by WMMAs
      const int k0n = (kt + 1) << 5;
      a0 = *(const uint4v*)&A[arow0 + k0n + ca];
      a1 = *(const uint4v*)&A[arow1 + k0n + ca];
      b0 = *(const uint4v*)&B[(long)(k0n + rb0) * N + bn + cb];
      b1 = *(const uint4v*)&B[(long)(k0n + rb1) * N + bn + cb];
    }

    v16bf bf[4];
#pragma unroll
    for (int nt = 0; nt < 4; ++nt) {
      int n = wn * 64 + nt * 16 + l16;
#pragma unroll
      for (int v = 0; v < 8; ++v)
#pragma unroll
        for (int s = 0; s < 2; ++s)
          bf[nt][2 * v + s] = sB[cur][koff16(v, half, s) * 136 + n];
    }
#pragma unroll
    for (int mt = 0; mt < 2; ++mt) {
      int m = wm * 32 + mt * 16 + l16;
      v16bf af;
#pragma unroll
      for (int v = 0; v < 8; ++v)
#pragma unroll
        for (int s = 0; s < 2; ++s)
          af[2 * v + s] = sA[cur][m * 40 + koff16(v, half, s)];
#pragma unroll
      for (int nt = 0; nt < 4; ++nt)
        acc[mt][nt] = __builtin_amdgcn_wmma_f32_16x16x32_bf16(
            false, af, false, bf[nt], (short)0, acc[mt][nt], false, false);
    }

    if (more) {
      const int nxt = cur ^ 1;
      *(uint4v*)&sA[nxt][ra0 * 40 + ca] = a0;
      *(uint4v*)&sA[nxt][ra1 * 40 + ca] = a1;
      *(uint4v*)&sB[nxt][rb0 * 136 + cb] = b0;
      *(uint4v*)&sB[nxt][rb1 * 136 + cb] = b1;
      __syncthreads();
    }
  }

#pragma unroll
  for (int mt = 0; mt < 2; ++mt)
#pragma unroll
    for (int nt = 0; nt < 4; ++nt) {
      long n = bn + wn * 64 + nt * 16 + l16;
      float biasv = bias[n];
#pragma unroll
      for (int r = 0; r < 8; ++r) {
        long m = bm + wm * 32 + mt * 16 + half * 8 + r;
        float val = acc[mt][nt][r] + biasv;
        if constexpr (OUT_BF16)
          ((bf16_t*)C)[m * (long)N + n] = (bf16_t)val;
        else
          ((float*)C)[m * (long)N + n] = val;
      }
    }
}

// ---------------------------------------------------------------------------
// Causal normalized attention:
//   ctx[t] = (1/exp(n_t)) * sum_{s<=t} (q_t . k_s) v_s      per (batch, head)
// block = (b*H+h, 128-query tile), 8 waves, 16 query rows each.
// All causal skips are SCALAR (readfirstlane wave id) so every WMMA executes
// with full EXEC as the ISA requires; skips become s_cmp/s_cbranch.
// ---------------------------------------------------------------------------
__global__ __launch_bounds__(256) void attn_wmma(
    const bf16_t* __restrict__ qk,    // (ROWS, 2048): q cols [0,1024), k [1024,2048)
    const bf16_t* __restrict__ vbuf,  // (ROWS, 1024)
    const float* __restrict__ nbuf,   // (ROWS, 16)
    bf16_t* __restrict__ ctx) {       // (ROWS, 1024)
  __shared__ bf16_t sQ[128 * 72];
  __shared__ bf16_t sK[64 * 72];
  __shared__ bf16_t sV[64 * 72];
  __shared__ bf16_t sS[8][16 * 72];   // per-wave score scratch (layout shuffle)
  __shared__ float sNs[128];

  const int tid = threadIdx.x;
  const int lane = tid & 31;
  // wave-uniform scalar: all causal-skip branches must be SALU branches
  const int wave = __builtin_amdgcn_readfirstlane(tid >> 5);
  const int half = lane >> 4, l16 = lane & 15;
  const int bh = blockIdx.x;
  const int b = bh >> 4, h = bh & 15;
  const int qb = blockIdx.y * 128;
  const long rowbase = (long)b * SEQ;

  // stage Q tile (128x64) once; reused across all key tiles
#pragma unroll
  for (int i = 0; i < 4; ++i) {
    int idx = tid + i * 256;
    int r = idx >> 3, c = (idx & 7) << 3;
    *(uint4v*)&sQ[r * 72 + c] =
        *(const uint4v*)&qk[(rowbase + qb + r) * 2048 + h * 64 + c];
  }
  if (tid < 128) sNs[tid] = __expf(-nbuf[(rowbase + qb + tid) * 16 + h]);

  const int q0 = qb + wave * 16;   // scalar
  const int tmax = q0 + 15;        // scalar
  const v8f vzero = {0.f, 0.f, 0.f, 0.f, 0.f, 0.f, 0.f, 0.f};
  v8f acc[4];
#pragma unroll
  for (int nt = 0; nt < 4; ++nt) acc[nt] = vzero;

  const int pr = tid >> 3, pc = (tid & 7) << 3;  // prefetch chunk coords
  const int nkt = (qb >> 6) + 2;  // 64-key tiles covering keys [0, qb+128)
  for (int jt = 0; jt < nkt; ++jt) {
    const int j0 = jt * 64;
    __syncthreads();
#pragma unroll
    for (int i = 0; i < 2; ++i) {   // stage K and V tiles (64x64 each)
      int idx = tid + i * 256;
      int r = idx >> 3, c = (idx & 7) << 3;
      *(uint4v*)&sK[r * 72 + c] =
          *(const uint4v*)&qk[(rowbase + j0 + r) * 2048 + 1024 + h * 64 + c];
      *(uint4v*)&sV[r * 72 + c] =
          *(const uint4v*)&vbuf[(rowbase + j0 + r) * 1024 + h * 64 + c];
    }
    __syncthreads();

    // prefetch next K/V tiles into cache while this one is computed
    if (jt + 1 < nkt) {
      __builtin_prefetch(
          &qk[(rowbase + j0 + 64 + pr) * 2048 + 1024 + h * 64 + pc], 0, 1);
      __builtin_prefetch(
          &vbuf[(rowbase + j0 + 64 + pr) * 1024 + h * 64 + pc], 0, 1);
    }

    if (j0 > tmax) continue;  // scalar: fully masked for this wave's rows

    // scores(16x64) = Q_tile @ K_tile^T (skip fully-masked 16-key sub-tiles)
    v8f sc[4];
#pragma unroll
    for (int nt = 0; nt < 4; ++nt) sc[nt] = vzero;
#pragma unroll
    for (int d0 = 0; d0 < 64; d0 += 32) {
      v16bf aq;
#pragma unroll
      for (int v = 0; v < 8; ++v)
#pragma unroll
        for (int s = 0; s < 2; ++s)
          aq[2 * v + s] = sQ[(wave * 16 + l16) * 72 + d0 + koff16(v, half, s)];
#pragma unroll
      for (int nt = 0; nt < 4; ++nt) {
        if (j0 + nt * 16 > tmax) continue;  // scalar causal skip
        v16bf bk;  // B[k=d][n=s_local] = K[s_local][d] (transpose via LDS)
#pragma unroll
        for (int v = 0; v < 8; ++v)
#pragma unroll
          for (int s = 0; s < 2; ++s)
            bk[2 * v + s] = sK[(nt * 16 + l16) * 72 + d0 + koff16(v, half, s)];
        sc[nt] = __builtin_amdgcn_wmma_f32_16x16x32_bf16(
            false, aq, false, bk, (short)0, sc[nt], false, false);
      }
    }

    // causal mask, round to bf16, stash in per-wave LDS scratch (C->A layout)
#pragma unroll
    for (int nt = 0; nt < 4; ++nt) {
      const int smin = j0 + nt * 16;   // scalar
      const int soff = (half * 8) * 72 + nt * 16 + l16;
      if (smin + 15 <= q0) {            // scalar: fully unmasked sub-tile
#pragma unroll
        for (int r = 0; r < 8; ++r)
          sS[wave][soff + r * 72] = (bf16_t)sc[nt][r];
      } else if (smin > tmax) {         // scalar: fully masked sub-tile
#pragma unroll
        for (int r = 0; r < 8; ++r)
          sS[wave][soff + r * 72] = (bf16_t)0.0f;
      } else {                          // diagonal sub-tile: element mask
#pragma unroll
        for (int r = 0; r < 8; ++r) {
          int t = q0 + half * 8 + r;
          int s = smin + l16;
          float val = (s <= t) ? sc[nt][r] : 0.0f;
          sS[wave][soff + r * 72] = (bf16_t)val;
        }
      }
    }

    // ctx(16x64) += attn(16x64) @ V(64x64); skip fully-masked 32-key k-steps
#pragma unroll
    for (int d0 = 0; d0 < 64; d0 += 32) {
      if (j0 + d0 > tmax) continue;     // scalar causal skip
      v16bf aa;
#pragma unroll
      for (int v = 0; v < 8; ++v)
#pragma unroll
        for (int s = 0; s < 2; ++s)
          aa[2 * v + s] = sS[wave][l16 * 72 + d0 + koff16(v, half, s)];
#pragma unroll
      for (int nt = 0; nt < 4; ++nt) {
        v16bf bvf;  // B[k=s_local][n=d] = V[s_local][d]
#pragma unroll
        for (int v = 0; v < 8; ++v)
#pragma unroll
          for (int s = 0; s < 2; ++s)
            bvf[2 * v + s] = sV[(d0 + koff16(v, half, s)) * 72 + nt * 16 + l16];
        acc[nt] = __builtin_amdgcn_wmma_f32_16x16x32_bf16(
            false, aa, false, bvf, (short)0, acc[nt], false, false);
      }
    }
  }

  // epilogue: scale by 1/exp(n_t), store bf16 ctx
#pragma unroll
  for (int nt = 0; nt < 4; ++nt)
#pragma unroll
    for (int r = 0; r < 8; ++r) {
      int row = wave * 16 + half * 8 + r;
      int t = qb + row;
      float val = acc[nt][r] * sNs[row];
      ctx[(rowbase + t) * 1024 + h * 64 + nt * 16 + l16] = (bf16_t)val;
    }
}

// ---------------------------------------------------------------------------
// launch
// ---------------------------------------------------------------------------
extern "C" void kernel_launch(void* const* d_in, const int* in_sizes, int n_in,
                              void* d_out, int out_size, void* d_ws,
                              size_t ws_size, hipStream_t stream) {
  (void)in_sizes; (void)n_in; (void)out_size; (void)ws_size;
  const float* x   = (const float*)d_in[0];
  const float* Wqk = (const float*)d_in[1];
  const float* bqk = (const float*)d_in[2];
  const float* Wv  = (const float*)d_in[3];
  const float* bv  = (const float*)d_in[4];
  const float* Wn  = (const float*)d_in[5];
  const float* bn  = (const float*)d_in[6];
  const float* Wo  = (const float*)d_in[7];
  const float* bo  = (const float*)d_in[8];
  float* out = (float*)d_out;

  char* ws = (char*)d_ws;
  bf16_t* xb   = (bf16_t*)ws;  ws += (size_t)ROWS * 1024 * sizeof(bf16_t);
  bf16_t* Wqkb = (bf16_t*)ws;  ws += (size_t)1024 * 2048 * sizeof(bf16_t);
  bf16_t* Wvb  = (bf16_t*)ws;  ws += (size_t)1024 * 1024 * sizeof(bf16_t);
  bf16_t* Wob  = (bf16_t*)ws;  ws += (size_t)1024 * 1024 * sizeof(bf16_t);
  bf16_t* qkb  = (bf16_t*)ws;  ws += (size_t)ROWS * 2048 * sizeof(bf16_t);
  bf16_t* vbw  = (bf16_t*)ws;  ws += (size_t)ROWS * 1024 * sizeof(bf16_t);
  bf16_t* ctxb = (bf16_t*)ws;  ws += (size_t)ROWS * 1024 * sizeof(bf16_t);
  float*  nbuf = (float*)ws;   ws += (size_t)ROWS * NUM_HEADS * sizeof(float);

  // 1) bf16 conversions
  {
    int n;
    n = ROWS * 1024;
    f32_to_bf16<<<(n + 255) / 256, 256, 0, stream>>>(x, xb, n);
    n = 1024 * 2048;
    f32_to_bf16<<<(n + 255) / 256, 256, 0, stream>>>(Wqk, Wqkb, n);
    n = 1024 * 1024;
    f32_to_bf16<<<(n + 255) / 256, 256, 0, stream>>>(Wv, Wvb, n);
    n = 1024 * 1024;
    f32_to_bf16<<<(n + 255) / 256, 256, 0, stream>>>(Wo, Wob, n);
  }

  // 2) tiny n-projection (f32)
  proj_n<<<(ROWS * NUM_HEADS + 255) / 256, 256, 0, stream>>>(x, Wn, bn, nbuf);

  // 3) qk = x @ Wqk + bqk  (4096 x 2048 x 1024)
  gemm_bias_wmma<true><<<dim3(ROWS / 128, 2048 / 128), 256, 0, stream>>>(
      xb, Wqkb, bqk, qkb, ROWS, 2048, 1024);

  // 4) v = x @ Wv + bv     (4096 x 1024 x 1024)
  gemm_bias_wmma<true><<<dim3(ROWS / 128, 1024 / 128), 256, 0, stream>>>(
      xb, Wvb, bv, vbw, ROWS, 1024, 1024);

  // 5) causal normalized attention -> ctx (bf16)
  attn_wmma<<<dim3(BATCH * NUM_HEADS, SEQ / 128), 256, 0, stream>>>(
      qkb, vbw, nbuf, ctxb);

  // 6) out = ctx @ Wo + bo (f32 output)
  gemm_bias_wmma<false><<<dim3(ROWS / 128, 1024 / 128), 256, 0, stream>>>(
      ctxb, Wob, bo, out, ROWS, 1024, 1024);
}
